// QuantizedLinearActMXFP4_635655160008
// MI455X (gfx1250) — matrix operand order repaired and checked by reading, not compiled, not run on hardware
//
#include <hip/hip_runtime.h>
#include <hip/hip_bf16.h>
#include <math.h>

typedef _Float16 half8  __attribute__((ext_vector_type(8)));
typedef _Float16 half16 __attribute__((ext_vector_type(16)));
typedef float    float8 __attribute__((ext_vector_type(8)));
typedef int      int4v  __attribute__((ext_vector_type(4)));

#define GLOBAL_AS __attribute__((address_space(1)))
#define LDS_AS    __attribute__((address_space(3)))

// ---- CDNA5 async global->LDS copy (ASYNCcnt-tracked, no VGPR round-trip) ----
// Probe-confirmed signature: (int4 as1*, int4 as3*, imm i32 offset, imm i32 cpol)
#if __has_builtin(__builtin_amdgcn_global_load_async_to_lds_b128)
#define ASYNC_COPY_B128(gptr, lptr)                                            \
    __builtin_amdgcn_global_load_async_to_lds_b128(                            \
        (GLOBAL_AS int4v*)(void*)(gptr), (LDS_AS int4v*)(void*)(lptr), 0, 0)
#else
#define ASYNC_COPY_B128(gptr, lptr)                                            \
    asm volatile("global_load_async_to_lds_b128 %0, %1, off" ::                \
                     "v"((unsigned)(uintptr_t)(lptr)),                         \
                     "v"((unsigned long long)(uintptr_t)(gptr))                \
                 : "memory")
#endif

#if __has_builtin(__builtin_amdgcn_s_wait_asynccnt)
#define WAIT_ASYNC0() __builtin_amdgcn_s_wait_asynccnt(0)
#else
#define WAIT_ASYNC0() asm volatile("s_wait_asynccnt 0x0" ::: "memory")
#endif

// ---------------------------------------------------------------------------
// MXFP4 fake-quant: one lane per element, one wave32 per 32-element block.
// amax -> shared exp2 scale (fp4 emax=2), delta tweak 0.5*tanh(4*delta) as
// exp2(d), E2M1 round-to-nearest-even (rintf == jnp.round), y = scale * q,
// emitted as f16 for the WMMA GEMM (f32 accumulate downstream).
// ---------------------------------------------------------------------------
__global__ void mxfp4_quant_f16(const float* __restrict__ x,
                                const float* __restrict__ delta,
                                int per_block_delta,
                                _Float16* __restrict__ out,
                                long long n) {
    long long idx = (long long)blockIdx.x * 256 + threadIdx.x;
    float v = (idx < n) ? x[idx] : 0.0f;
    float a = fabsf(v);
    a = fmaxf(a, __shfl_xor(a, 1));
    a = fmaxf(a, __shfl_xor(a, 2));
    a = fmaxf(a, __shfl_xor(a, 4));
    a = fmaxf(a, __shfl_xor(a, 8));
    a = fmaxf(a, __shfl_xor(a, 16));
    if (idx >= n) return;
    long long blk = idx >> 5;
    float dlt = per_block_delta ? delta[blk] : delta[0];
    float d = 0.5f * tanhf(4.0f * dlt);                 // range_scale*tanh(beta*delta)
    float e = floorf(log2f(fmaxf(a, 1e-30f)));
    float scale = (a > 0.0f) ? exp2f(e - 2.0f) : 1.0f;  // fp4 emax = 2
    scale *= exp2f(d);
    float xs  = v / scale;
    float axs = fminf(fabsf(xs), 6.0f);                 // FP4_MAX
    float eq  = fminf(fmaxf(floorf(log2f(fmaxf(axs, 1e-30f))), 0.0f), 2.0f);
    float step = exp2f(eq - 1.0f);                      // 0.5 / 1 / 2
    float q = rintf(axs / step) * step;                 // RNE == jnp.round
    float y = copysignf(q, xs) * scale;
    out[idx] = (_Float16)y;
}

// ---------------------------------------------------------------------------
// WMMA GEMM: Y[M,N] = Xq[M,K] * Wq[N,K]^T + bias[N]   (f16 in, f32 accum/out)
// 256 threads = 8 waves, 128x128 output tile; waveM = wave&3 (32 rows),
// waveN = wave>>2 (64 cols) -> 8 v_wmma_f32_16x16x32_f16 accumulators/wave.
// K staged in 64-wide chunks via async global->LDS copies into a double
// buffer (2 x 36 KB); one barrier per chunk; fills of chunk k+1 overlap the
// WMMAs of chunk k. Rows padded 64->72 halves: the 16-row b128 fragment
// reads tile all 64 LDS banks exactly once.
// ---------------------------------------------------------------------------
#define LDS_PITCH 72

__global__ void __launch_bounds__(256)
wmma_gemm_f16(const _Float16* __restrict__ Xq,
              const _Float16* __restrict__ Wq,
              const float* __restrict__ bias,
              float* __restrict__ Y,
              int M, int N, int K) {
    __shared__ __align__(16) _Float16 ldsA[2][128 * LDS_PITCH];
    __shared__ __align__(16) _Float16 ldsB[2][128 * LDS_PITCH];

    const int tid   = threadIdx.x;
    const int wave  = tid >> 5;
    const int lane  = tid & 31;
    const int lrow  = lane & 15;
    const int lhi   = lane >> 4;          // 0 | 1
    const int waveM = wave & 3;           // 4 waves along M
    const int waveN = wave >> 2;          // 2 waves along N
    const int mBase = blockIdx.y * 128;
    const int nBase = blockIdx.x * 128;

    float8 acc[2][4];
#pragma unroll
    for (int mt = 0; mt < 2; ++mt)
#pragma unroll
        for (int nt = 0; nt < 4; ++nt)
            acc[mt][nt] = (float8)(0.0f);

    // Async-fill one 128x64 A tile + B tile (b128 per lane, 4 issues each).
    auto stage = [&](int kc, int buf) {
#pragma unroll
        for (int i = 0; i < 4; ++i) {
            int idx = tid + i * 256;      // 0..1023
            int row = idx >> 3;
            int g   = (idx & 7) * 8;
            ASYNC_COPY_B128(&Xq[(size_t)(mBase + row) * K + kc + g],
                            &ldsA[buf][row * LDS_PITCH + g]);
            ASYNC_COPY_B128(&Wq[(size_t)(nBase + row) * K + kc + g],
                            &ldsB[buf][row * LDS_PITCH + g]);
        }
    };

    stage(0, 0);
    int buf = 0;
    for (int kc = 0; kc < K; kc += 64, buf ^= 1) {
        WAIT_ASYNC0();        // my fills of `buf` landed in LDS
        __syncthreads();      // everyone's landed; everyone done reading buf^1
        if (kc + 64 < K) stage(kc + 64, buf ^ 1);

#pragma unroll
        for (int ks = 0; ks < 64; ks += 32) {
            // A fragment (16x32 f16, ISA 7.12.2): lane<16 -> row=lane,
            // K = {ks..ks+7, ks+16..ks+23}; lane>=16 -> row=lane-16, K += 8.
            half16 af[2];
#pragma unroll
            for (int mt = 0; mt < 2; ++mt) {
                const _Float16* p =
                    &ldsA[buf][(waveM * 32 + mt * 16 + lrow) * LDS_PITCH + ks + lhi * 8];
                half8 lo = *(const half8*)p;
                half8 hi = *(const half8*)(p + 16);
                af[mt] = __builtin_shufflevector(lo, hi, 0, 1, 2, 3, 4, 5, 6, 7,
                                                 8, 9, 10, 11, 12, 13, 14, 15);
            }
            // B fragment (32x16 f16): lane<16 -> col=lane, K = {ks..ks+7, ks+16..ks+23};
            // lane>=16 -> col=lane-16, K += 8. Column n of B is
            // row n of Wq (K-contiguous): plain b128 LDS reads.
            half16 bf[4];
#pragma unroll
            for (int nt = 0; nt < 4; ++nt) {
                const _Float16* p =
                    &ldsB[buf][(waveN * 64 + nt * 16 + lrow) * LDS_PITCH + ks + lhi * 8];
                half8 lo = *(const half8*)p;
                half8 hi = *(const half8*)(p + 16);
                bf[nt] = __builtin_shufflevector(lo, hi, 0, 1, 2, 3, 4, 5, 6, 7,
                                                 8, 9, 10, 11, 12, 13, 14, 15);
            }
#pragma unroll
            for (int mt = 0; mt < 2; ++mt)
#pragma unroll
                for (int nt = 0; nt < 4; ++nt)
                    acc[mt][nt] = __builtin_amdgcn_wmma_f32_16x16x32_f16(
                        false, af[mt], false, bf[nt],
                        (short)0, acc[mt][nt], false, false);
        }
    }

    // C layout: VGPR r, lane l -> M = r + 8*(l>>4), N = l&15. Add bias, store.
#pragma unroll
    for (int nt = 0; nt < 4; ++nt) {
        int n = nBase + waveN * 64 + nt * 16 + lrow;
        float bv = bias[n];
#pragma unroll
        for (int mt = 0; mt < 2; ++mt) {
            int m0 = mBase + waveM * 32 + mt * 16 + lhi * 8;
#pragma unroll
            for (int r = 0; r < 8; ++r)
                Y[(size_t)(m0 + r) * N + n] = acc[mt][nt][r] + bv;
        }
    }
}

// ---------------------------------------------------------------------------
// inputs: 0:x[B,S,K] f32  1:weight[N,K] f32  2:bias[N] f32
//         3:delta_w[N,K/32] f32  4:delta_act[1] f32
// out: y[B,S,N] f32.  ws: x_q f16 (M*K*2 B) + w_q f16 (N*K*2 B) ~= 66 MB.
// ---------------------------------------------------------------------------
extern "C" void kernel_launch(void* const* d_in, const int* in_sizes, int n_in,
                              void* d_out, int out_size, void* d_ws, size_t ws_size,
                              hipStream_t stream) {
    const float* x        = (const float*)d_in[0];
    const float* w        = (const float*)d_in[1];
    const float* bias     = (const float*)d_in[2];
    const float* delta_w  = (const float*)d_in[3];
    const float* delta_a  = (const float*)d_in[4];

    const long long xN = in_sizes[0];
    const long long wN = in_sizes[1];
    const int K = (int)(32LL * wN / in_sizes[3]);   // 32-wide blocks along K
    const int N = (int)(wN / K);
    const int M = (int)(xN / K);

    _Float16* xq = (_Float16*)d_ws;
    _Float16* wq = (_Float16*)((char*)d_ws + (size_t)xN * sizeof(_Float16));

    mxfp4_quant_f16<<<(unsigned)((xN + 255) / 256), 256, 0, stream>>>(
        x, delta_a, /*per_block=*/0, xq, xN);
    mxfp4_quant_f16<<<(unsigned)((wN + 255) / 256), 256, 0, stream>>>(
        w, delta_w, /*per_block=*/1, wq, wN);

    dim3 grid(N / 128, M / 128);
    wmma_gemm_f16<<<grid, 256, 0, stream>>>(xq, wq, bias, (float*)d_out, M, N, K);
}
